// GriffinLlamaMLP_36266703848196
// MI455X (gfx1250) — matrix operand order repaired and checked by reading, not compiled, not run on hardware
//
#include <hip/hip_runtime.h>

// ---------------------------------------------------------------------------
// Types for CDNA5 WMMA (wave32, v_wmma_f32_16x16x32_bf16)
// ---------------------------------------------------------------------------
typedef __bf16 bf16;
typedef __attribute__((ext_vector_type(16))) __bf16 v16bf;
typedef __attribute__((ext_vector_type(8)))  float  v8f;

union Frag { v16bf v; uint4 q[2]; };

#define BM 128
#define BN 128
#define BK 32          // K-depth of fallback (f32-input) GEMM
#define BKA 64         // K-depth of async bf16 GEMM (2 wmma K-steps)

#define S_DIM 2048
#define H_DIM 4096
#define I_DIM 11008
#define K_ZERO 5504    // int(0.5 * 11008)
#define SP_THREADS 256
#define CPT 43         // 256 * 43 == 11008 exactly

// exact pointee type expected by the async-to-LDS builtins:
//   param0: v4i addrspace(1)*, param1: v4i addrspace(3)*
typedef int v4i __attribute__((__vector_size__(16)));
#define AS_GLOBAL(p) ((__attribute__((address_space(1))) v4i*)(p))
#define AS_LDS(p)    ((__attribute__((address_space(3))) v4i*)(p))

// ---------------------------------------------------------------------------
// f32 -> bf16 bulk conversion (grid-stride, float4 -> 4x bf16)
// ---------------------------------------------------------------------------
__global__ __launch_bounds__(256) void cvt_f32_bf16(
    const float* __restrict__ in, bf16* __restrict__ out, long n4)
{
  long i = (long)blockIdx.x * blockDim.x + threadIdx.x;
  const long stride = (long)gridDim.x * blockDim.x;
  for (; i < n4; i += stride) {
    const float4 f = ((const float4*)in)[i];
    union { bf16 h[4]; uint2 u; } o;
    o.h[0] = (bf16)f.x; o.h[1] = (bf16)f.y;
    o.h[2] = (bf16)f.z; o.h[3] = (bf16)f.w;
    ((uint2*)out)[i] = o.u;
  }
}

// ---------------------------------------------------------------------------
// Async-pipelined bf16 GEMM: C[M,N] = A[M,K] * B[N,K]^T  (A,B bf16, C f32).
// Tiles staged global->LDS with GLOBAL_LOAD_ASYNC_TO_LDS_B128 (ASYNCcnt),
// double-buffered. M%128==0, N%128==0, K%64==0 for all calls here.
// ---------------------------------------------------------------------------
__global__ __launch_bounds__(256) void gemm_bf16_wmma_async(
    const bf16* __restrict__ A, const bf16* __restrict__ B,
    float* __restrict__ C, int M, int N, int K, int apply_silu)
{
  __shared__ __attribute__((aligned(16))) bf16 sA[2][BM * BKA];  // 2 x 16 KB
  __shared__ __attribute__((aligned(16))) bf16 sB[2][BN * BKA];  // 2 x 16 KB

  const int t    = threadIdx.x;
  const int lane = t & 31;
  const int wave = t >> 5;        // 0..7
  const int wm   = wave >> 2;     // 0..1  -> 64 rows each
  const int wn   = wave & 3;      // 0..3  -> 32 cols each

  const int m0 = blockIdx.y * BM;
  const int n0 = blockIdx.x * BN;

  // staging: each thread owns 64 B of a row (4 x b128 async ops) per matrix
  const int lrow  = t >> 1;           // 0..127
  const int lcolE = (t & 1) << 5;     // 0 or 32 bf16 elements (64 B)

  v8f acc[4][2];
#pragma unroll
  for (int i = 0; i < 4; ++i)
#pragma unroll
    for (int j = 0; j < 2; ++j) {
      v8f z = {0.f, 0.f, 0.f, 0.f, 0.f, 0.f, 0.f, 0.f};
      acc[i][j] = z;
    }

  const int ktiles = K / BKA;

  // issue one tile's async copies into LDS buffer `buf`
  auto issue = [&](int kt, int buf) {
    const int k0 = kt * BKA;
    const bf16* ga = A + (size_t)(m0 + lrow) * K + k0 + lcolE;
    const bf16* gb = B + (size_t)(n0 + lrow) * K + k0 + lcolE;
    bf16* la = &sA[buf][lrow * BKA + lcolE];
    bf16* lb = &sB[buf][lrow * BKA + lcolE];
    // INST_OFFSET advances both the global and LDS address (ISA 15.18)
    __builtin_amdgcn_global_load_async_to_lds_b128(AS_GLOBAL(ga), AS_LDS(la), 0,  0);
    __builtin_amdgcn_global_load_async_to_lds_b128(AS_GLOBAL(ga), AS_LDS(la), 16, 0);
    __builtin_amdgcn_global_load_async_to_lds_b128(AS_GLOBAL(ga), AS_LDS(la), 32, 0);
    __builtin_amdgcn_global_load_async_to_lds_b128(AS_GLOBAL(ga), AS_LDS(la), 48, 0);
    __builtin_amdgcn_global_load_async_to_lds_b128(AS_GLOBAL(gb), AS_LDS(lb), 0,  0);
    __builtin_amdgcn_global_load_async_to_lds_b128(AS_GLOBAL(gb), AS_LDS(lb), 16, 0);
    __builtin_amdgcn_global_load_async_to_lds_b128(AS_GLOBAL(gb), AS_LDS(lb), 32, 0);
    __builtin_amdgcn_global_load_async_to_lds_b128(AS_GLOBAL(gb), AS_LDS(lb), 48, 0);
  };

  issue(0, 0);

  const int akoff = (lane >> 4) << 4;   // A frag K sub-offset: 0 or 16 elems
  const int bkoff = (lane >> 4) << 4;   // B frag K sub-offset: 0 or 16 elems

  for (int kt = 0; kt < ktiles; ++kt) {
    const int buf = kt & 1;
    __builtin_amdgcn_s_wait_asynccnt(0);   // my writes into sX[buf] done
    __syncthreads();                       // everyone's writes visible
    if (kt + 1 < ktiles) issue(kt + 1, buf ^ 1);

#pragma unroll
    for (int ks = 0; ks < BKA; ks += 32) {
      // A fragments (16-bit A 16x32): lanes 0-15 hold K 0..15;
      // lanes 16-31 hold K 16..31; M = lane&15.
      Frag a[4], b[2];
#pragma unroll
      for (int i = 0; i < 4; ++i) {
        const int r = wm * 64 + i * 16 + (lane & 15);
        const uint4* p = (const uint4*)(&sA[buf][r * BKA + ks + akoff]);
        a[i].q[0] = p[0];
        a[i].q[1] = p[1];
      }
      // B fragments: lane n&15 holds 16 contiguous K of W-row; lanes 0-15 ->
      // K 0..15, lanes 16-31 -> K 16..31.
#pragma unroll
      for (int j = 0; j < 2; ++j) {
        const int r = wn * 32 + j * 16 + (lane & 15);
        const uint4* p = (const uint4*)(&sB[buf][r * BKA + ks + bkoff]);
        b[j].q[0] = p[0];
        b[j].q[1] = p[1];
      }
#pragma unroll
      for (int i = 0; i < 4; ++i)
#pragma unroll
        for (int j = 0; j < 2; ++j)
          acc[i][j] = __builtin_amdgcn_wmma_f32_16x16x32_bf16(
              false, a[i].v, false, b[j].v, (short)0, acc[i][j], false, false);
    }
    __syncthreads();   // done reading sX[buf]; next-next issue may overwrite
  }

  // Epilogue. C/D layout: lane l, vgpr r -> M = r + (l>=16 ? 8 : 0), N = l&15.
#pragma unroll
  for (int i = 0; i < 4; ++i) {
    const int row0 = m0 + wm * 64 + i * 16 + ((lane >> 4) << 3);
#pragma unroll
    for (int j = 0; j < 2; ++j) {
      const int col = n0 + wn * 32 + j * 16 + (lane & 15);
#pragma unroll
      for (int r = 0; r < 8; ++r) {
        float v = acc[i][j][r];
        if (apply_silu) v = v / (1.0f + __expf(-v));
        C[(size_t)(row0 + r) * N + col] = v;
      }
    }
  }
}

// ---------------------------------------------------------------------------
// Fallback GEMM: f32 inputs, in-register bf16 conversion.
// ---------------------------------------------------------------------------
__device__ inline uint4 cvt8(const float* __restrict__ p) {
  union { bf16 h[8]; uint4 q; } o;
#pragma unroll
  for (int i = 0; i < 8; ++i) o.h[i] = (bf16)p[i];
  return o.q;
}

__global__ __launch_bounds__(256) void gemm_bf16_wmma_f32in(
    const float* __restrict__ A, const float* __restrict__ B,
    float* __restrict__ C, int M, int N, int K, int apply_silu)
{
  __shared__ __attribute__((aligned(16))) bf16 sA[BM * BK];
  __shared__ __attribute__((aligned(16))) bf16 sB[BN * BK];

  const int t    = threadIdx.x;
  const int lane = t & 31;
  const int wave = t >> 5;
  const int wm   = wave >> 2;
  const int wn   = wave & 3;

  const int m0 = blockIdx.y * BM;
  const int n0 = blockIdx.x * BN;

  const int lrow = t >> 1;
  const int lcol = (t & 1) << 4;

  v8f acc[4][2];
#pragma unroll
  for (int i = 0; i < 4; ++i)
#pragma unroll
    for (int j = 0; j < 2; ++j) {
      v8f z = {0.f, 0.f, 0.f, 0.f, 0.f, 0.f, 0.f, 0.f};
      acc[i][j] = z;
    }

  const int ktiles = K / BK;
  for (int kt = 0; kt < ktiles; ++kt) {
    const int k0 = kt * BK;
    const float* ga = A + (size_t)(m0 + lrow) * K + k0 + lcol;
    const float* gb = B + (size_t)(n0 + lrow) * K + k0 + lcol;

    uint4* la = (uint4*)(sA + lrow * BK + lcol);
    uint4* lb = (uint4*)(sB + lrow * BK + lcol);
    la[0] = cvt8(ga);  la[1] = cvt8(ga + 8);
    lb[0] = cvt8(gb);  lb[1] = cvt8(gb + 8);

    if (kt + 1 < ktiles) {
      __builtin_prefetch(ga + BK, 0, 0);
      __builtin_prefetch(gb + BK, 0, 0);
    }
    __syncthreads();

    Frag a[4], b[2];
    const int akoff = (lane >> 4) << 4;
    const int bkoff = (lane >> 4) << 4;
#pragma unroll
    for (int i = 0; i < 4; ++i) {
      const int r = wm * 64 + i * 16 + (lane & 15);
      const uint4* p = (const uint4*)(sA + r * BK + akoff);
      a[i].q[0] = p[0];
      a[i].q[1] = p[1];
    }
#pragma unroll
    for (int j = 0; j < 2; ++j) {
      const int r = wn * 32 + j * 16 + (lane & 15);
      const uint4* p = (const uint4*)(sB + r * BK + bkoff);
      b[j].q[0] = p[0];
      b[j].q[1] = p[1];
    }
#pragma unroll
    for (int i = 0; i < 4; ++i)
#pragma unroll
      for (int j = 0; j < 2; ++j)
        acc[i][j] = __builtin_amdgcn_wmma_f32_16x16x32_bf16(
            false, a[i].v, false, b[j].v, (short)0, acc[i][j], false, false);
    __syncthreads();
  }

#pragma unroll
  for (int i = 0; i < 4; ++i) {
    const int row0 = m0 + wm * 64 + i * 16 + ((lane >> 4) << 3);
#pragma unroll
    for (int j = 0; j < 2; ++j) {
      const int col = n0 + wn * 32 + j * 16 + (lane & 15);
#pragma unroll
      for (int r = 0; r < 8; ++r) {
        float v = acc[i][j][r];
        if (apply_silu) v = v / (1.0f + __expf(-v));
        C[(size_t)(row0 + r) * N + col] = v;
      }
    }
  }
}

// ---------------------------------------------------------------------------
// Per-token exact top-k zeroing + up-product. Radix-select (4 x 8-bit MSB
// passes) over as_uint(|g|) finds the K_ZERO-th smallest magnitude, with
// deterministic tie handling. Output: bf16 into sb if sb != nullptr, else
// f32 in place over g.
// ---------------------------------------------------------------------------
__global__ __launch_bounds__(SP_THREADS) void sparsify_topk(
    float* __restrict__ g, const float* __restrict__ u, bf16* __restrict__ sb)
{
  const int row = blockIdx.x;
  const int tid = threadIdx.x;
  float*       gr = g + (size_t)row * I_DIM;
  const float* ur = u + (size_t)row * I_DIM;
  bf16*        sr = sb ? (sb + (size_t)row * I_DIM) : (bf16*)nullptr;

  __shared__ unsigned hist[256];
  __shared__ int      cnts[256];
  __shared__ unsigned s_prefix;
  __shared__ int      s_kk;

  const int base_i = tid * CPT;

  if (tid == 0) { s_prefix = 0u; s_kk = K_ZERO; }
  __syncthreads();

  for (int pass = 0; pass < 4; ++pass) {
    const int shift = 24 - 8 * pass;
    hist[tid] = 0u;
    __syncthreads();

    const unsigned prefix = s_prefix;
    const unsigned hm = (pass == 0) ? 0u : (0xFFFFFFFFu << (shift + 8));
    for (int e = 0; e < CPT; ++e) {
      const unsigned key = __float_as_uint(fabsf(gr[base_i + e]));
      if ((key & hm) == prefix)
        atomicAdd(&hist[(key >> shift) & 255u], 1u);
    }
    __syncthreads();

    if (tid == 0) {
      unsigned cum = 0;
      const unsigned kk = (unsigned)s_kk;
      unsigned b = 0;
      for (; b < 256u; ++b) {
        const unsigned h = hist[b];
        if (cum + h >= kk) break;
        cum += h;
      }
      s_prefix = prefix | (b << shift);
      s_kk = (int)(kk - cum);
    }
    __syncthreads();
  }

  const unsigned T    = s_prefix;
  const int      need = s_kk;

  int eq = 0;
  for (int e = 0; e < CPT; ++e) {
    const unsigned key = __float_as_uint(fabsf(gr[base_i + e]));
    eq += (key == T);
  }
  cnts[tid] = eq;
  __syncthreads();
  if (tid == 0) {
    int run = 0;
    for (int i = 0; i < 256; ++i) { const int c = cnts[i]; cnts[i] = run; run += c; }
  }
  __syncthreads();
  const int base_eq = cnts[tid];

  int seen = 0;
  for (int e = 0; e < CPT; ++e) {
    const int idx = base_i + e;
    const float gv = gr[idx];
    const unsigned key = __float_as_uint(fabsf(gv));
    float val;
    if (key < T) {
      val = 0.0f;
    } else if (key == T) {
      val = (base_eq + seen < need) ? 0.0f : gv * ur[idx];
      ++seen;
    } else {
      val = gv * ur[idx];
    }
    if (sr) sr[idx] = (bf16)val;
    else    gr[idx] = val;
  }
}

// ---------------------------------------------------------------------------
// Launch. Big-ws path: convert everything to bf16 once, run async WMMA GEMMs.
// Small-ws fallback: round-1 convert-in-kernel GEMMs (180 MB workspace).
// ---------------------------------------------------------------------------
extern "C" void kernel_launch(void* const* d_in, const int* in_sizes, int n_in,
                              void* d_out, int out_size, void* d_ws, size_t ws_size,
                              hipStream_t stream)
{
  (void)in_sizes; (void)n_in; (void)out_size;
  const float* x  = (const float*)d_in[0];   // [2048, 4096]
  const float* wg = (const float*)d_in[1];   // [11008, 4096]
  const float* wu = (const float*)d_in[2];   // [11008, 4096]
  const float* wd = (const float*)d_in[3];   // [4096, 11008]
  float* out = (float*)d_out;                // [2048, 4096]

  const size_t SH = (size_t)S_DIM * H_DIM;   // 8,388,608
  const size_t IH = (size_t)I_DIM * H_DIM;   // 45,088,768
  const size_t SI = (size_t)S_DIM * I_DIM;   // 22,544,384

  auto al = [](size_t v) { return (v + 255) & ~(size_t)255; };
  const size_t xbB  = al(SH * 2);
  const size_t wbB  = al(IH * 2);
  const size_t gB   = al(SI * 4);
  const size_t sbB  = al(SI * 2);
  const size_t need_big = xbB + 3 * wbB + 2 * gB + sbB;   // ~513 MB

  const dim3 blk(256);

  if (ws_size >= need_big) {
    char* p = (char*)d_ws;
    bf16* xb  = (bf16*)p;               p += xbB;
    bf16* wgb = (bf16*)p;               p += wbB;
    bf16* wub = (bf16*)p;               p += wbB;
    bf16* wdb = (bf16*)p;               p += wbB;
    float* g  = (float*)p;              p += gB;
    float* u  = (float*)p;              p += gB;
    bf16* sb  = (bf16*)p;

    cvt_f32_bf16<<<dim3(2048), blk, 0, stream>>>(x,  xb,  (long)(SH / 4));
    cvt_f32_bf16<<<dim3(2048), blk, 0, stream>>>(wg, wgb, (long)(IH / 4));
    cvt_f32_bf16<<<dim3(2048), blk, 0, stream>>>(wu, wub, (long)(IH / 4));
    cvt_f32_bf16<<<dim3(2048), blk, 0, stream>>>(wd, wdb, (long)(IH / 4));

    gemm_bf16_wmma_async<<<dim3(I_DIM / BN, S_DIM / BM), blk, 0, stream>>>(
        xb, wgb, g, S_DIM, I_DIM, H_DIM, /*silu=*/1);
    gemm_bf16_wmma_async<<<dim3(I_DIM / BN, S_DIM / BM), blk, 0, stream>>>(
        xb, wub, u, S_DIM, I_DIM, H_DIM, /*silu=*/0);
    sparsify_topk<<<dim3(S_DIM), dim3(SP_THREADS), 0, stream>>>(g, u, sb);
    gemm_bf16_wmma_async<<<dim3(H_DIM / BN, S_DIM / BM), blk, 0, stream>>>(
        sb, wdb, out, S_DIM, H_DIM, I_DIM, /*silu=*/0);
  } else {
    float* g = (float*)d_ws;                       // reused as s (f32) in place
    float* u = (float*)((char*)d_ws + gB);

    gemm_bf16_wmma_f32in<<<dim3(I_DIM / BN, S_DIM / BM), blk, 0, stream>>>(
        x, wg, g, S_DIM, I_DIM, H_DIM, /*silu=*/1);
    gemm_bf16_wmma_f32in<<<dim3(I_DIM / BN, S_DIM / BM), blk, 0, stream>>>(
        x, wu, u, S_DIM, I_DIM, H_DIM, /*silu=*/0);
    sparsify_topk<<<dim3(S_DIM), dim3(SP_THREADS), 0, stream>>>(g, u, (bf16*)nullptr);
    gemm_bf16_wmma_f32in<<<dim3(H_DIM / BN, S_DIM / BM), blk, 0, stream>>>(
        g, wd, out, S_DIM, H_DIM, I_DIM, /*silu=*/0);
  }
}
